// TH_LSTM_65489661329493
// MI455X (gfx1250) — matrix operand order repaired and checked by reading, not compiled, not run on hardware
//
#include <hip/hip_runtime.h>
#include <hip/hip_bf16.h>
#include <math.h>

// Problem dims
#define BB 4096
#define TT 16
#define FF 64      // F (feature incl. time channel)
#define HH 512
#define DD 15
#define NTOT 2560  // 4H (gates) + H (time-aware gate pre-act)
#define KCAT 576   // 64 (x+t) + 512 (h)

typedef _Float16 v16h __attribute__((ext_vector_type(16)));
typedef _Float16 v8h  __attribute__((ext_vector_type(8)));
typedef float    v8f  __attribute__((ext_vector_type(8)));

__device__ __forceinline__ float sigm(float x) { return 1.0f / (1.0f + expf(-x)); }

// ---------------- WMMA fragment loaders (contiguous k = 16*half + s, same for A and B, wave32) -------
// A 16x32 f16: lanes 0-15 -> M=lane, K=0..15; lanes 16-31 -> K=16..31
__device__ __forceinline__ v16h load_a_frag(const _Float16* __restrict__ A, int lda,
                                            int row0, int k0, int lane) {
  const int half = lane >> 4, m = lane & 15;
  const _Float16* base = A + (size_t)(row0 + m) * lda + k0 + 16 * half;
  v8h lo = *(const v8h*)(base);        // K = k0 + 16*half + [0..8)
  v8h hi = *(const v8h*)(base + 8);    // K = k0 + 16*half + [8..16)
  v16h r;
#pragma unroll
  for (int i = 0; i < 8; ++i) { r[i] = lo[i]; r[i + 8] = hi[i]; }
  return r;
}

// B 32x16 f16 from column-major storage Bt[N][K]: lanes 0-15 -> N=lane, K=k0+0..15;
// lanes 16-31 -> K=k0+16..31
__device__ __forceinline__ v16h load_bt_frag(const _Float16* __restrict__ Bt, int ldbt,
                                             int n0, int k0, int lane) {
  const int half = lane >> 4, n = lane & 15;
  const _Float16* base = Bt + (size_t)(n0 + n) * ldbt + k0 + 16 * half;
  v8h lo = *(const v8h*)(base);
  v8h hi = *(const v8h*)(base + 8);
  v16h r;
#pragma unroll
  for (int i = 0; i < 8; ++i) { r[i] = lo[i]; r[i + 8] = hi[i]; }
  return r;
}

// ---------------- Generic WMMA GEMM: C[M,N](f32) = A[M,K](f16) * B (f16, stored B^T) ----
// grid = (N/128, M/128), block = 256 (8 waves, 4x2), wave tile = 32x64 (2x4 WMMA tiles)
__global__ void __launch_bounds__(256)
wmma_gemm(const _Float16* __restrict__ A, int lda,
          const _Float16* __restrict__ Bt, int ldbt,
          float* __restrict__ C, int ldc, int K) {
  const int lane = threadIdx.x & 31;
  const int wave = threadIdx.x >> 5;
  const int m_base = blockIdx.y * 128 + (wave >> 1) * 32;
  const int n_base = blockIdx.x * 128 + (wave & 1) * 64;

  v8f acc[2][4] = {};

  for (int k0 = 0; k0 < K; k0 += 32) {
    v16h a0 = load_a_frag(A, lda, m_base, k0, lane);
    v16h a1 = load_a_frag(A, lda, m_base + 16, k0, lane);
#pragma unroll
    for (int j = 0; j < 4; ++j) {
      v16h b = load_bt_frag(Bt, ldbt, n_base + j * 16, k0, lane);
      acc[0][j] = __builtin_amdgcn_wmma_f32_16x16x32_f16(
          false, a0, false, b, (short)0, acc[0][j], false, false);
      acc[1][j] = __builtin_amdgcn_wmma_f32_16x16x32_f16(
          false, a1, false, b, (short)0, acc[1][j], false, false);
    }
  }
  const int half = lane >> 4, n = lane & 15;
#pragma unroll
  for (int i = 0; i < 2; ++i)
#pragma unroll
    for (int j = 0; j < 4; ++j)
#pragma unroll
      for (int r = 0; r < 8; ++r)
        C[(size_t)(m_base + i * 16 + r + 8 * half) * ldc + n_base + j * 16 + n] =
            acc[i][j][r];
}

// ---------------- Setup kernels -------------------------------------------------------
// WcatT[NTOT][KCAT]: transposed concat of [row0 = (Wt gates | 0 | 0); Wx|WxT ; Uh|0]
__global__ void build_wcatT(const float* __restrict__ Wx, const float* __restrict__ Uh,
                            const float* __restrict__ Wt, const float* __restrict__ WxT,
                            _Float16* __restrict__ WcatT) {
  int idx = blockIdx.x * 256 + threadIdx.x;           // over NTOT*KCAT
  int n = idx / KCAT, k = idx % KCAT;
  float v;
  if (n < 2048) {                                      // gate columns
    if (k == 0)       v = (n < 1536) ? Wt[n] : 0.0f;   // t -> i,f,o only
    else if (k < 64)  v = Wx[(size_t)(k - 1) * 2048 + n];
    else              v = Uh[(size_t)(k - 64) * 2048 + n];
  } else {                                             // time-aware gate columns
    int j = n - 2048;
    if (k == 0)       v = 0.0f;
    else if (k < 64)  v = WxT[(size_t)(k - 1) * HH + j];
    else              v = 0.0f;
  }
  WcatT[idx] = (_Float16)v;
}

// WcombT[HH][1024] = transpose of [We; Wg] stacked on K
__global__ void build_wcombT(const float* __restrict__ We, const float* __restrict__ Wg,
                             _Float16* __restrict__ WcombT) {
  int idx = blockIdx.x * 256 + threadIdx.x;           // over HH*1024
  int n = idx >> 10, k = idx & 1023;
  float v = (k < HH) ? We[(size_t)k * HH + n] : Wg[(size_t)(k - HH) * HH + n];
  WcombT[idx] = (_Float16)v;
}

// out[N][K] = (f16) W[K][N]
__global__ void transpose_to_f16(const float* __restrict__ W, _Float16* __restrict__ O,
                                 int K, int N) {
  int idx = blockIdx.x * 256 + threadIdx.x;           // over N*K
  int n = idx / K, k = idx % K;
  O[idx] = (_Float16)W[(size_t)k * N + n];
}

__global__ void init_state(const float* __restrict__ hc_init,
                           _Float16* __restrict__ Abuf, float* __restrict__ c) {
  int idx = blockIdx.x * 256 + threadIdx.x;           // over BB*HH
  int b = idx >> 9, j = idx & 511;
  Abuf[(size_t)b * KCAT + FF + j] = (_Float16)hc_init[j];
  c[idx] = hc_init[j];
}

__global__ void init_row_k(const float* __restrict__ init_proj, float* __restrict__ row) {
  int h = blockIdx.x * 256 + threadIdx.x;             // 512
  float s = 0.0f;
  for (int f = 0; f < FF; ++f) s += init_proj[(size_t)f * HH + h];
  row[h] = s;
}

__global__ void row_wa_k(const float* __restrict__ row, const float* __restrict__ Wa,
                         float* __restrict__ rowwa) {
  int h = blockIdx.x * 256 + threadIdx.x;             // 512
  float s = 0.0f;
  for (int k = 0; k < HH; ++k) s += row[k] * Wa[(size_t)k * HH + h];
  rowwa[h] = s;
}

__global__ void fill_mem(const float* __restrict__ row, const float* __restrict__ rowwa,
                         _Float16* __restrict__ mem16, float* __restrict__ memWa) {
  size_t idx = (size_t)blockIdx.x * 256 + threadIdx.x;  // over BB*DD*HH
  int h = (int)(idx & 511);
  mem16[idx] = (_Float16)row[h];
  memWa[idx] = rowwa[h];
}

// ---------------- Per-step kernels ----------------------------------------------------
__global__ void pack_x(const float* __restrict__ inputs, int t, _Float16* __restrict__ Abuf) {
  int idx = blockIdx.x * 256 + threadIdx.x;           // over BB*FF
  int b = idx >> 6, f = idx & 63;
  Abuf[(size_t)b * KCAT + f] = (_Float16)inputs[((size_t)b * TT + t) * FF + f];
}

__global__ void gates_kernel(const float* __restrict__ Z, const float* __restrict__ inputs,
                             int t, const float* __restrict__ b4,
                             const float* __restrict__ bT, const float* __restrict__ WtT,
                             float* __restrict__ c, _Float16* __restrict__ A2) {
  int idx = blockIdx.x * 256 + threadIdx.x;           // over BB*HH
  int b = idx >> 9, j = idx & 511;
  size_t zb = (size_t)b * NTOT;
  float zi = Z[zb + j]        + b4[j];
  float zf = Z[zb + 512 + j]  + b4[512 + j];
  float zo = Z[zb + 1024 + j] + b4[1024 + j];
  float zc = Z[zb + 1536 + j] + b4[1536 + j];
  float g  = Z[zb + 2048 + j] + bT[j];
  float tv = inputs[((size_t)b * TT + t) * FF];        // time-delta channel
  float i_ = sigm(zi), f_ = sigm(zf), o_ = sigm(zo);
  float ch = tanhf(zc);
  float Tg = sigm(g + sigm(tv * WtT[j]));
  float cn = f_ * c[idx] + i_ * ch + Tg * ch;
  c[idx] = cn;
  A2[(size_t)b * 1024 + HH + j] = (_Float16)(o_ * tanhf(cn));  // h_cur
}

__global__ void attn_kernel(const float* __restrict__ memWa,
                            const _Float16* __restrict__ mem16,
                            const float* __restrict__ hWb, const float* __restrict__ ba,
                            const float* __restrict__ va, _Float16* __restrict__ A2) {
  int b = blockIdx.x;
  int tid = threadIdx.x;                               // 256
  __shared__ float e[DD];
  __shared__ float alpha[DD];
  if (tid < DD) e[tid] = 0.0f;
  __syncthreads();
  float hb[2], vv[2];
#pragma unroll
  for (int s = 0; s < 2; ++s) {
    int h = tid + s * 256;
    hb[s] = hWb[(size_t)b * HH + h] + ba[h];
    vv[s] = va[h];
  }
  for (int d = 0; d < DD; ++d) {
    float p = 0.0f;
#pragma unroll
    for (int s = 0; s < 2; ++s) {
      int h = tid + s * 256;
      p += tanhf(memWa[((size_t)b * DD + d) * HH + h] + hb[s]) * vv[s];
    }
    atomicAdd(&e[d], p);                               // LDS ds_add_f32
  }
  __syncthreads();
  if (tid == 0) {
    float mx = e[0];
    for (int d = 1; d < DD; ++d) mx = fmaxf(mx, e[d]);
    float s = 0.0f;
    for (int d = 0; d < DD; ++d) { alpha[d] = expf(e[d] - mx); s += alpha[d]; }
    float inv = 1.0f / s;
    for (int d = 0; d < DD; ++d) alpha[d] *= inv;
  }
  __syncthreads();
#pragma unroll
  for (int s = 0; s < 2; ++s) {
    int h = tid + s * 256;
    float cx = 0.0f;
    for (int d = 0; d < DD; ++d)
      cx += alpha[d] * (float)mem16[((size_t)b * DD + d) * HH + h];
    A2[(size_t)b * 1024 + h] = (_Float16)cx;           // ctx
  }
}

__global__ void combine_kernel(const float* __restrict__ comb, const float* __restrict__ bh,
                               _Float16* __restrict__ Abuf, _Float16* __restrict__ mem16,
                               int slot) {
  int idx = blockIdx.x * 256 + threadIdx.x;            // over BB*HH
  int b = idx >> 9, j = idx & 511;
  float v = tanhf(comb[idx] + bh[j]);
  Abuf[(size_t)b * KCAT + FF + j] = (_Float16)v;       // h for next step / head
  mem16[((size_t)b * DD + slot) * HH + j] = (_Float16)v;
}

__global__ void head_kernel(const float* __restrict__ Zh, const float* __restrict__ b1,
                            const float* __restrict__ W2, const float* __restrict__ b2,
                            const float* __restrict__ W3, const float* __restrict__ b3,
                            float* __restrict__ out) {
  int b = blockIdx.x;
  int tid = threadIdx.x;                               // 32 (one wave)
  __shared__ float hd1[256];
  __shared__ float hd2[32];
  for (int i = tid; i < 256; i += 32) hd1[i] = fmaxf(Zh[(size_t)b * 256 + i] + b1[i], 0.0f);
  __syncthreads();
  float a = b2[tid];
  for (int k = 0; k < 256; ++k) a += hd1[k] * W2[k * 32 + tid];
  hd2[tid] = fmaxf(a, 0.0f);
  __syncthreads();
  if (tid == 0) {
    float l0 = b3[0], l1 = b3[1];
    for (int j = 0; j < 32; ++j) { l0 += hd2[j] * W3[j * 2]; l1 += hd2[j] * W3[j * 2 + 1]; }
    float m = fmaxf(l0, l1);
    float e0 = expf(l0 - m), e1 = expf(l1 - m), inv = 1.0f / (e0 + e1);
    out[b * 2] = e0 * inv;
    out[b * 2 + 1] = e1 * inv;
  }
}

// ---------------- Host orchestration --------------------------------------------------
extern "C" void kernel_launch(void* const* d_in, const int* in_sizes, int n_in,
                              void* d_out, int out_size, void* d_ws, size_t ws_size,
                              hipStream_t stream) {
  (void)in_sizes; (void)n_in; (void)out_size; (void)ws_size;
  const float* inputs    = (const float*)d_in[0];
  const float* init_proj = (const float*)d_in[1];
  const float* hc_init   = (const float*)d_in[2];
  const float* Wx  = (const float*)d_in[3];
  const float* Uh  = (const float*)d_in[4];
  const float* Wt  = (const float*)d_in[5];
  const float* bb  = (const float*)d_in[6];
  const float* WxT = (const float*)d_in[7];
  const float* WtT = (const float*)d_in[8];
  const float* bT  = (const float*)d_in[9];
  const float* Wa  = (const float*)d_in[10];
  const float* Wb  = (const float*)d_in[11];
  const float* va  = (const float*)d_in[12];
  const float* ba  = (const float*)d_in[13];
  const float* We  = (const float*)d_in[14];
  const float* Wg  = (const float*)d_in[15];
  const float* bh  = (const float*)d_in[16];
  const float* W1  = (const float*)d_in[17];
  const float* b1  = (const float*)d_in[18];
  const float* W2  = (const float*)d_in[19];
  const float* b2  = (const float*)d_in[20];
  const float* W3  = (const float*)d_in[21];
  const float* b3  = (const float*)d_in[22];
  float* out = (float*)d_out;

  char* ws = (char*)d_ws;
  size_t off = 0;
  auto alloc = [&](size_t bytes) -> char* {
    char* p = ws + off;
    off += (bytes + 255) & ~(size_t)255;
    return p;
  };
  _Float16* WcatT  = (_Float16*)alloc((size_t)NTOT * KCAT * 2);
  _Float16* WcombT = (_Float16*)alloc((size_t)HH * 1024 * 2);
  _Float16* WbT    = (_Float16*)alloc((size_t)HH * HH * 2);
  _Float16* WaT    = (_Float16*)alloc((size_t)HH * HH * 2);
  _Float16* W1T    = (_Float16*)alloc((size_t)256 * HH * 2);
  _Float16* Abuf   = (_Float16*)alloc((size_t)BB * KCAT * 2);   // [x(64) | h(512)] f16
  _Float16* A2     = (_Float16*)alloc((size_t)BB * 1024 * 2);   // [ctx(512) | h_cur(512)]
  float* Z     = (float*)alloc((size_t)BB * NTOT * 4);
  float* cst   = (float*)alloc((size_t)BB * HH * 4);
  float* hWb   = (float*)alloc((size_t)BB * HH * 4);
  float* comb  = (float*)alloc((size_t)BB * HH * 4);
  _Float16* mem16 = (_Float16*)alloc((size_t)BB * DD * HH * 2); // mem ring (f16)
  float* memWa = (float*)alloc((size_t)BB * DD * HH * 4);       // mem@Wa ring (f32)
  float* initrow = (float*)alloc(HH * 4);
  float* rowwa   = (float*)alloc(HH * 4);

  // ---- setup (runs every call: deterministic state rebuild) ----
  build_wcatT<<<(NTOT * KCAT) / 256, 256, 0, stream>>>(Wx, Uh, Wt, WxT, WcatT);
  build_wcombT<<<(HH * 1024) / 256, 256, 0, stream>>>(We, Wg, WcombT);
  transpose_to_f16<<<(HH * HH) / 256, 256, 0, stream>>>(Wb, WbT, HH, HH);
  transpose_to_f16<<<(HH * HH) / 256, 256, 0, stream>>>(Wa, WaT, HH, HH);
  transpose_to_f16<<<(HH * 256) / 256, 256, 0, stream>>>(W1, W1T, HH, 256);
  init_state<<<(BB * HH) / 256, 256, 0, stream>>>(hc_init, Abuf, cst);
  init_row_k<<<2, 256, 0, stream>>>(init_proj, initrow);
  row_wa_k<<<2, 256, 0, stream>>>(initrow, Wa, rowwa);
  fill_mem<<<(BB * DD * HH) / 256, 256, 0, stream>>>(initrow, rowwa, mem16, memWa);

  // ---- sequential scan over T ----
  for (int t = 0; t < TT; ++t) {
    pack_x<<<(BB * FF) / 256, 256, 0, stream>>>(inputs, t, Abuf);
    // Z = [x|h] @ [Wbig; Uh]  (M=4096, N=2560, K=576)
    wmma_gemm<<<dim3(NTOT / 128, BB / 128), 256, 0, stream>>>(Abuf, KCAT, WcatT, KCAT,
                                                              Z, NTOT, KCAT);
    gates_kernel<<<(BB * HH) / 256, 256, 0, stream>>>(Z, inputs, t, bb, bT, WtT, cst, A2);
    // hWb = h_cur @ Wb
    wmma_gemm<<<dim3(HH / 128, BB / 128), 256, 0, stream>>>(A2 + HH, 1024, WbT, HH,
                                                            hWb, HH, HH);
    attn_kernel<<<BB, 256, 0, stream>>>(memWa, mem16, hWb, ba, va, A2);
    // comb = ctx@We + h_cur@Wg  (K=1024 concat)
    wmma_gemm<<<dim3(HH / 128, BB / 128), 256, 0, stream>>>(A2, 1024, WcombT, 1024,
                                                            comb, HH, 1024);
    combine_kernel<<<(BB * HH) / 256, 256, 0, stream>>>(comb, bh, Abuf, mem16, t % DD);
    // memWa[slot] = h_new @ Wa  (direct write into ring via ldc trick)
    wmma_gemm<<<dim3(HH / 128, BB / 128), 256, 0, stream>>>(
        Abuf + FF, KCAT, WaT, HH, memWa + (size_t)(t % DD) * HH, DD * HH, HH);
  }

  // ---- classifier head ----
  wmma_gemm<<<dim3(256 / 128, BB / 128), 256, 0, stream>>>(Abuf + FF, KCAT, W1T, HH,
                                                           Z, 256, HH);
  head_kernel<<<BB, 32, 0, stream>>>(Z, b1, W2, b2, W3, b3, out);
}
